// GCN_85890755986035
// MI455X (gfx1250) — matrix operand order repaired and checked by reading, not compiled, not run on hardware
//
#include <hip/hip_runtime.h>
#include <stdint.h>

// ---------------------------------------------------------------------------
// GCN layer on MI455X (gfx1250):
//   agg = A @ X          A: [16384,16384] f32, X: [16384,256] f32
//   out = relu(agg@W + b)                  W: [256,256] f32,  b: [256] f32
//
// Roofline: A streamed once (1 GiB) -> 46us @ 23.3 TB/s.  fp32 WMMA (K=4)
// would be ~10x compute-bound; bf16 WMMA (K=32) with fp32 accumulate is
// balanced with HBM.  So: convert A/X/W -> bf16, accumulate f32.
// ---------------------------------------------------------------------------

typedef __attribute__((ext_vector_type(4)))  float   f32x4;
typedef __attribute__((ext_vector_type(8)))  float   v8f;
typedef __attribute__((ext_vector_type(8)))  __bf16  v8bf;
typedef __attribute__((ext_vector_type(16))) __bf16  v16bf;

#define NNODES 16384
#define CDIM   256
#define FDIM   256
#define MTILE  128                 // rows of A per workgroup (8 waves x 16)
#define KSTEPS (NNODES / 32)       // 512 bf16-WMMA K-steps
#define XTILE_BF 8192              // 32 k x 256 n bf16 = 16 KB per tile
#define AG_STRIDE 264              // padded agg row (528 B: 16B-aligned, bank-friendly)

// XOR-swizzle on 16-byte LDS chunks: spreads the per-lane 64B-strided
// B-fragment reads across banks (8-way -> ~2-way conflicts). Invertible.
__device__ __forceinline__ int swz(int c) { return c ^ ((c >> 2) & 15); }

// CDNA5 async global->LDS copy (16B per lane), tracked by ASYNCcnt.
__device__ __forceinline__ void async_cp16(uint32_t lds_addr, const void* gptr) {
  asm volatile("global_load_async_to_lds_b128 %0, %1, off"
               :: "v"(lds_addr), "v"((uint64_t)(uintptr_t)gptr)
               : "memory");
}
__device__ __forceinline__ void wait_async0() {
  asm volatile("s_wait_asynccnt 0x0" ::: "memory");
}

__device__ __forceinline__ v16bf cat16(v8bf lo, v8bf hi) {
  return __builtin_shufflevector(lo, hi, 0,1,2,3,4,5,6,7,8,9,10,11,12,13,14,15);
}

// ---------------------------------------------------------------------------
// Pre-pass: fp32 row-major -> bf16 "B-fragment blocked" layout:
//   dst[kc*8192 + n*32 + kk] = bf16(src[(kc*32+kk)*256 + n])
// Used for both X (16384x256) and W (256x256).  Writes fully coalesced.
// ---------------------------------------------------------------------------
__global__ void cvt_blocked_bf16(const float* __restrict__ src,
                                 __bf16* __restrict__ dst) {
  int t  = blockIdx.x * 256 + threadIdx.x;
  int kc = t >> 13;        // / 8192
  int r  = t & 8191;
  int n  = r >> 5;
  int kk = r & 31;
  dst[t] = (__bf16)src[(size_t)(kc * 32 + kk) * 256 + n];
}

// ---------------------------------------------------------------------------
// Main fused kernel: one WG = 128 rows, 8 waves, wave strip = 16 x 256.
// ---------------------------------------------------------------------------
__global__ __launch_bounds__(256, 1)
void gcn_fused(const float* __restrict__ A, const __bf16* __restrict__ xbf,
               const __bf16* __restrict__ wbf, const float* __restrict__ bias,
               float* __restrict__ out) {
  // Phase 1 uses smem as 2 x 16KB X-tile double buffer; phase 2 overlays it
  // with the per-wave bf16 agg strips (safe: separated by the loop barrier).
  __shared__ __align__(16) __bf16 smem[8 * 16 * AG_STRIDE];  // 33792 bf16 = 66 KB

  const int tid  = threadIdx.x;
  const int w    = tid >> 5;        // wave 0..7
  const int l    = tid & 31;        // lane
  const int u    = l & 15;
  const int half = l >> 4;          // 0: lanes 0-15, 1: lanes 16-31
  const int rowBase = blockIdx.x * MTILE;

  v8f acc[16];
#pragma unroll
  for (int t = 0; t < 16; ++t) acc[t] = (v8f){0.f,0.f,0.f,0.f,0.f,0.f,0.f,0.f};

  // A fragment source: lane (u,half) owns row rowBase+16w+u, K-bytes at
  // half*8 within each 32-wide K chunk (matches WMMA 16-bit A layout).
  const float* arow = A + (size_t)(rowBase + w * 16 + u) * NNODES + half * 8;

  // -------- prologue: async-stage X tile 0 into buffer 0 --------
#pragma unroll
  for (int i = 0; i < 4; ++i) {
    int c = tid + 256 * i;                       // 16B chunk id within tile
    async_cp16((uint32_t)(uintptr_t)&smem[swz(c) * 8], xbf + c * 8);
  }
  wait_async0();
  __syncthreads();

  // -------- phase 1: agg = A @ X --------
  for (int kc = 0; kc < KSTEPS; ++kc) {
    const int cur = kc & 1;
    if (kc + 1 < KSTEPS) {                       // prefetch next tile (async)
      const __bf16* g = xbf + (size_t)(kc + 1) * XTILE_BF;
#pragma unroll
      for (int i = 0; i < 4; ++i) {
        int c = tid + 256 * i;
        async_cp16((uint32_t)(uintptr_t)&smem[(cur ^ 1) * XTILE_BF + swz(c) * 8],
                   g + c * 8);
      }
    }

    // A fragment: 16 fp32 from global, convert to bf16 in registers.
    const f32x4* ap = (const f32x4*)(arow + kc * 32);
    f32x4 q0 = ap[0], q1 = ap[1], q2 = ap[4], q3 = ap[5];
    v16bf af;
    af[0]=(__bf16)q0.x; af[1]=(__bf16)q0.y; af[2]=(__bf16)q0.z; af[3]=(__bf16)q0.w;
    af[4]=(__bf16)q1.x; af[5]=(__bf16)q1.y; af[6]=(__bf16)q1.z; af[7]=(__bf16)q1.w;
    af[8]=(__bf16)q2.x; af[9]=(__bf16)q2.y; af[10]=(__bf16)q2.z; af[11]=(__bf16)q2.w;
    af[12]=(__bf16)q3.x; af[13]=(__bf16)q3.y; af[14]=(__bf16)q3.z; af[15]=(__bf16)q3.w;

    const __bf16* buf = smem + cur * XTILE_BF;
#pragma unroll
    for (int nt = 0; nt < 16; ++nt) {
      int c = (nt * 16 + u) * 4 + half;          // lane's B-frag chunks
      v8bf lo = *(const v8bf*)(buf + swz(c) * 8);
      v8bf hi = *(const v8bf*)(buf + swz(c + 2) * 8);
      acc[nt] = __builtin_amdgcn_wmma_f32_16x16x32_bf16(
          false, af, false, cat16(lo, hi), (short)0, acc[nt], false, false);
    }

    wait_async0();        // next buffer landed
    __syncthreads();      // all waves done reading cur; writes visible
  }

  // -------- restage agg strip (bf16, A-fragment order) into LDS --------
  // Overlays the X buffers; safe past the loop's final barrier.  Each wave
  // writes and later reads only its own strip (LDS ops in-order per wave).
  __bf16* ag = smem;
#pragma unroll
  for (int nt = 0; nt < 16; ++nt)
#pragma unroll
    for (int r = 0; r < 8; ++r)
      ag[(w * 16 + r + 8 * half) * AG_STRIDE + nt * 16 + u] = (__bf16)acc[nt][r];

  // -------- phase 2: out = relu(agg @ W + b), K = 256 --------
  v8f acc2[16];
#pragma unroll
  for (int t = 0; t < 16; ++t) acc2[t] = (v8f){0.f,0.f,0.f,0.f,0.f,0.f,0.f,0.f};

#pragma unroll
  for (int kc2 = 0; kc2 < 8; ++kc2) {
    int a0 = (w * 16 + u) * AG_STRIDE + kc2 * 32 + half * 8;
    v16bf a2 = cat16(*(const v8bf*)(ag + a0), *(const v8bf*)(ag + a0 + 16));
#pragma unroll
    for (int nt = 0; nt < 16; ++nt) {
      const __bf16* wp = wbf + kc2 * 8192 + (nt * 16 + u) * 32 + half * 8;
      v16bf bf = cat16(*(const v8bf*)(wp), *(const v8bf*)(wp + 16));
      acc2[nt] = __builtin_amdgcn_wmma_f32_16x16x32_bf16(
          false, a2, false, bf, (short)0, acc2[nt], false, false);
    }
  }

  // bias + relu + store (fp32, row-major [16384,256])
#pragma unroll
  for (int nt = 0; nt < 16; ++nt) {
    float bb = bias[nt * 16 + u];
#pragma unroll
    for (int r = 0; r < 8; ++r) {
      float v = acc2[nt][r] + bb;
      v = v > 0.f ? v : 0.f;
      out[(size_t)(rowBase + w * 16 + r + 8 * half) * FDIM + nt * 16 + u] = v;
    }
  }
}

// ---------------------------------------------------------------------------
extern "C" void kernel_launch(void* const* d_in, const int* in_sizes, int n_in,
                              void* d_out, int out_size, void* d_ws, size_t ws_size,
                              hipStream_t stream) {
  const float* A = (const float*)d_in[0];   // [16384,16384]
  const float* X = (const float*)d_in[1];   // [16384,256]
  const float* W = (const float*)d_in[2];   // [256,256]
  const float* b = (const float*)d_in[3];   // [256]
  float* out = (float*)d_out;

  // Workspace: bf16 X (8 MB) + bf16 W (128 KB), both in blocked B layout.
  __bf16* xbf = (__bf16*)d_ws;
  __bf16* wbf = xbf + (size_t)NNODES * CDIM;

  cvt_blocked_bf16<<<(NNODES * CDIM) / 256, 256, 0, stream>>>(X, xbf);
  cvt_blocked_bf16<<<(CDIM * FDIM) / 256, 256, 0, stream>>>(W, wbf);
  gcn_fused<<<NNODES / MTILE, 256, 0, stream>>>(A, xbf, wbf, b, out);
}
